// DeepSphereUNetCD_50740743635155
// MI455X (gfx1250) — compile-verified, matrix operand order checked by reading, not run on hardware
//
#include <hip/hip_runtime.h>

// ---------------------------------------------------------------------------
// DeepSphere U-Net (Chebyshev K=3) for MI455X / gfx1250.
// Dense channel GEMMs on bf16 WMMA (v_wmma_f32_16x16x32_bf16, f32 accum) with
// fragment-ready pre-swizzled weights: inner loop = 4x global_load_b128 +
// 1x v_wmma + prefetch. SpMM / BN / pooling plumbing stays fp32 (net is
// HBM-bound: ~13 GFLOP of GEMM vs hundreds of MB at 23.3 TB/s).
// ---------------------------------------------------------------------------

typedef __attribute__((ext_vector_type(16))) __bf16 v16bf;
typedef __attribute__((ext_vector_type(8)))  float  v8f;

static constexpr int kDeg = 8;   // edges per vertex (rows = repeat(arange(V), 8))
static constexpr int kB   = 4;   // batch

__device__ __forceinline__ unsigned short f2bf(float f) {
  unsigned int u = __float_as_uint(f);
  unsigned int r = 0x7FFFu + ((u >> 16) & 1u);     // round-to-nearest-even
  return (unsigned short)((u + r) >> 16);
}

// y[b,v,c] = alpha * sum_j vals[8v+j] * x[b, cols[8v+j], c]  (- sub[b,v,c])
__global__ void spmm_kernel(const float* __restrict__ x, const int* __restrict__ cols,
                            const float* __restrict__ vals, const float* __restrict__ sub,
                            float* __restrict__ y, int V, int C, float alpha) {
  long long idx = (long long)blockIdx.x * blockDim.x + threadIdx.x;
  long long total = (long long)kB * V * C;
  if (idx >= total) return;
  int c = (int)(idx % C);
  int v = (int)((idx / C) % V);
  int b = (int)(idx / ((long long)C * V));
  const float* xb = x + (long long)b * V * C;
  int e0 = v * kDeg;
  float acc = 0.f;
#pragma unroll
  for (int j = 0; j < kDeg; ++j) {
    int col = cols[e0 + j];
    acc += vals[e0 + j] * xb[(long long)col * C + c];
  }
  acc *= alpha;
  if (sub) acc -= sub[idx];
  y[idx] = acc;
}

// A[row, p*3+k] = xk[row, p] (bf16), zero pad columns [3P, Kp)
__global__ void pack_xs_kernel(const float* __restrict__ x0, const float* __restrict__ x1,
                               const float* __restrict__ x2, unsigned short* __restrict__ A,
                               int BV, int P, int Kp) {
  long long idx = (long long)blockIdx.x * blockDim.x + threadIdx.x;
  long long total = (long long)BV * Kp;
  if (idx >= total) return;
  int c = (int)(idx % Kp);
  long long row = idx / Kp;
  float v = 0.f;
  if (c < 3 * P) {
    int k = c % 3, p = c / 3;
    const float* src = (k == 0) ? x0 : (k == 1) ? x1 : x2;
    v = src[row * P + p];
  }
  A[idx] = f2bf(v);
}

// Pre-swizzle flat weight (faithful to torch weight.view(K*P, Q)) into
// fragment-ready bf16 layout matching the wave32 B-matrix VGPR layout:
//   WP[((nt*(Kp/32) + kt)*32 + lane)*16 + i]
//   lane = half*16 + lid : column q = nt*16 + lid
//   i<8 : K = kt*32 + half*8 + i ; i>=8 : K = kt*32 + 16 + half*8 + (i-8)
__global__ void prep_w_kernel(const float* __restrict__ w, unsigned short* __restrict__ WP,
                              int rows, int Q, int Kp) {
  long long idx = (long long)blockIdx.x * blockDim.x + threadIdx.x;
  long long total = (long long)Kp * Q;     // (Q/16)*(Kp/32)*32*16
  if (idx >= total) return;
  int i    = (int)(idx % 16);
  int lane = (int)((idx >> 4) % 32);
  int kt   = (int)((idx >> 9) % (Kp >> 5));
  int nt   = (int)(idx >> 9) / (Kp >> 5);
  int lid  = lane & 15;
  int half = lane >> 4;
  int k = kt * 32 + half * 8 + ((i < 8) ? i : (16 + i - 8));
  int q = nt * 16 + lid;
  float v = (k < rows) ? w[(long long)k * Q + q] : 0.f;
  WP[idx] = f2bf(v);
}

// Y[M,Q] = A[M,Kp] @ W[Kp,Q] + bias   (bf16 in, f32 out, wave-per-16x16-tile)
// WP is the pre-swizzled weight from prep_w_kernel.
__global__ __launch_bounds__(128)
void gemm_bias_kernel(const unsigned short* __restrict__ A, const unsigned short* __restrict__ WP,
                      const float* __restrict__ bias, float* __restrict__ Y,
                      int M, int Kp, int Q) {
  const int wave = threadIdx.x >> 5;      // 0..3, uniform in wave
  const int lane = threadIdx.x & 31;
  const int lid  = lane & 15;
  const int half = lane >> 4;             // 0: K 0-7/16-23, 1: K 8-15/24-31
  const int m0 = blockIdx.y * 16;
  const int n0 = (blockIdx.x * 4 + wave) * 16;
  if (n0 >= Q) return;                    // uniform per wave -> EXEC all-ones at WMMA

  union Frag { v16bf v; uint4 u[2]; unsigned short s[16]; };
  v8f acc = {};
  const unsigned short* Arow = A + (long long)(m0 + lid) * Kp + half * 8;
  const unsigned short* Bt   = WP + (((long long)(n0 >> 4) * (Kp >> 5)) * 32 + lane) * 16;
  const int nkt = Kp >> 5;

  for (int kt = 0; kt < nkt; ++kt) {
    const unsigned short* ap = Arow + kt * 32;
    const unsigned short* bp = Bt + (long long)kt * 32 * 16;
    __builtin_prefetch(ap + 128, 0, 0);            // next A tiles (speculative)
    __builtin_prefetch(bp + 32 * 16 * 2, 0, 0);    // next-next B tile
    Frag a;
    a.u[0] = *reinterpret_cast<const uint4*>(ap);        // K +0..7   (this half)
    a.u[1] = *reinterpret_cast<const uint4*>(ap + 16);   // K +16..23 (this half)
    Frag b;
    b.u[0] = *reinterpret_cast<const uint4*>(bp);        // fragment-ready
    b.u[1] = *reinterpret_cast<const uint4*>(bp + 8);
    acc = __builtin_amdgcn_wmma_f32_16x16x32_bf16(false, a.v, false, b.v,
                                                  (short)0, acc, false, false);
  }
  const int col = n0 + lid;
  const float bv = bias[col];
#pragma unroll
  for (int r = 0; r < 8; ++r) {
    int mrow = m0 + half * 8 + r;         // lanes 0-15: M r, lanes 16-31: M r+8
    Y[(long long)mrow * Q + col] = acc[r] + bv;
  }
}

// training-mode BN stats over M rows per channel -> scale/shift
__global__ void bn_stats_kernel(const float* __restrict__ Y, const float* __restrict__ gamma,
                                const float* __restrict__ beta, float* __restrict__ scale,
                                float* __restrict__ shift, int M, int Q) {
  const int q = blockIdx.x;
  __shared__ float ssum[256], ssq[256];
  float s = 0.f, s2 = 0.f;
  for (int m = threadIdx.x; m < M; m += 256) {
    float v = Y[(long long)m * Q + q];
    s += v; s2 += v * v;
  }
  ssum[threadIdx.x] = s; ssq[threadIdx.x] = s2;
  __syncthreads();
  for (int st = 128; st > 0; st >>= 1) {
    if ((int)threadIdx.x < st) {
      ssum[threadIdx.x] += ssum[threadIdx.x + st];
      ssq[threadIdx.x]  += ssq[threadIdx.x + st];
    }
    __syncthreads();
  }
  if (threadIdx.x == 0) {
    float mean = ssum[0] / (float)M;
    float var  = ssq[0] / (float)M - mean * mean;
    float sc   = gamma[q] * rsqrtf(var + 1e-5f);
    scale[q] = sc;
    shift[q] = beta[q] - mean * sc;
  }
}

__global__ void bn_relu_kernel(float* __restrict__ Y, const float* __restrict__ scale,
                               const float* __restrict__ shift, long long total, int Q) {
  long long idx = (long long)blockIdx.x * blockDim.x + threadIdx.x;
  if (idx >= total) return;
  int q = (int)(idx % Q);
  float v = Y[idx] * scale[q] + shift[q];
  Y[idx] = fmaxf(v, 0.f);
}

// MaxPool1d(4) over vertices: y[b,vo,c] = max_{j<4} x[b,4vo+j,c]
__global__ void maxpool_kernel(const float* __restrict__ x, float* __restrict__ y,
                               int Vout, int C) {
  long long idx = (long long)blockIdx.x * blockDim.x + threadIdx.x;
  long long total = (long long)kB * Vout * C;
  if (idx >= total) return;
  int c  = (int)(idx % C);
  int vo = (int)((idx / C) % Vout);
  int b  = (int)(idx / ((long long)C * Vout));
  const float* base = x + (((long long)b * Vout + vo) * 4) * C + c;
  float m = base[0];
  m = fmaxf(m, base[C]);
  m = fmaxf(m, base[2 * C]);
  m = fmaxf(m, base[3 * C]);
  y[idx] = m;
}

// out[b,v,:] = concat(up[b, v/4, :C1], skip[b, v, :C2])
__global__ void upconcat_kernel(const float* __restrict__ up, const float* __restrict__ skip,
                                float* __restrict__ out, int V, int C1, int C2) {
  const int C = C1 + C2;
  long long idx = (long long)blockIdx.x * blockDim.x + threadIdx.x;
  long long total = (long long)kB * V * C;
  if (idx >= total) return;
  int c = (int)(idx % C);
  int v = (int)((idx / C) % V);
  int b = (int)(idx / ((long long)C * V));
  float val = (c < C1) ? up[((long long)b * (V / 4) + (v >> 2)) * C1 + c]
                       : skip[((long long)b * V + v) * C2 + (c - C1)];
  out[idx] = val;
}

// mu/logvar heads: y [B,V,16] -> out[0:BV]=mu, out[BV:2BV]=logvar
__global__ void heads_kernel(const float* __restrict__ y, const float* __restrict__ mu_w,
                             const float* __restrict__ mu_b, const float* __restrict__ lv_w,
                             const float* __restrict__ lv_b, float* __restrict__ out, int BV) {
  int idx = blockIdx.x * blockDim.x + threadIdx.x;
  if (idx >= BV) return;
  const float* yr = y + (long long)idx * 16;
  float m = 0.f, l = 0.f;
#pragma unroll
  for (int c = 0; c < 16; ++c) {
    m += yr[c] * mu_w[c];
    l += yr[c] * lv_w[c];
  }
  out[idx]      = m + mu_b[0];
  out[BV + idx] = l + lv_b[0];
}

// ---------------------------------------------------------------------------
// Host-side orchestration
// ---------------------------------------------------------------------------
struct Blk { const float *w1, *b1, *g1, *be1, *w2, *b2, *g2, *be2; };

static inline unsigned gblk(long long n) { return (unsigned)((n + 255) / 256); }

static void run_cbr(hipStream_t stream, const int* cols, const float* vals, int V,
                    const float* xin, int P, int Q,
                    const float* w, const float* bias, const float* g, const float* be,
                    float* x1b, float* x2b, unsigned short* xs, unsigned short* wf,
                    float* ss, float* yout) {
  const long long BV = (long long)kB * V;
  const long long ts = BV * P;
  spmm_kernel<<<gblk(ts), 256, 0, stream>>>(xin, cols, vals, nullptr, x1b, V, P, 1.0f);
  spmm_kernel<<<gblk(ts), 256, 0, stream>>>(x1b, cols, vals, xin, x2b, V, P, 2.0f);
  const int Kp = ((3 * P + 31) / 32) * 32;
  pack_xs_kernel<<<gblk(BV * Kp), 256, 0, stream>>>(xin, x1b, x2b, xs, (int)BV, P, Kp);
  prep_w_kernel<<<gblk((long long)Kp * Q), 256, 0, stream>>>(w, wf, 3 * P, Q, Kp);
  dim3 ggrid((unsigned)((Q + 63) / 64), (unsigned)(BV / 16));
  gemm_bias_kernel<<<ggrid, 128, 0, stream>>>(xs, wf, bias, yout, (int)BV, Kp, Q);
  bn_stats_kernel<<<Q, 256, 0, stream>>>(yout, g, be, ss, ss + Q, (int)BV, Q);
  bn_relu_kernel<<<gblk(BV * Q), 256, 0, stream>>>(yout, ss, ss + Q, BV * Q, Q);
}

extern "C" void kernel_launch(void* const* d_in, const int* in_sizes, int n_in,
                              void* d_out, int out_size, void* d_ws, size_t ws_size,
                              hipStream_t stream) {
  (void)in_sizes; (void)n_in; (void)out_size; (void)ws_size;

  const float* x = (const float*)d_in[0];
  const int*   cols3 = (const int*)d_in[2];  const float* vals3 = (const float*)d_in[3];
  const int*   cols2 = (const int*)d_in[5];  const float* vals2 = (const float*)d_in[6];
  const int*   cols1 = (const int*)d_in[8];  const float* vals1 = (const float*)d_in[9];
  const int*   cols0 = (const int*)d_in[11]; const float* vals0 = (const float*)d_in[12];

  int pi = 13;
  auto nblk = [&](void) {
    Blk bl;
    bl.w1 = (const float*)d_in[pi++]; bl.b1 = (const float*)d_in[pi++];
    bl.g1 = (const float*)d_in[pi++]; bl.be1 = (const float*)d_in[pi++];
    bl.w2 = (const float*)d_in[pi++]; bl.b2 = (const float*)d_in[pi++];
    bl.g2 = (const float*)d_in[pi++]; bl.be2 = (const float*)d_in[pi++];
    return bl;
  };
  Blk E3 = nblk(), E2 = nblk(), E1 = nblk(), Bb = nblk(), D1 = nblk(), D2 = nblk(), D3 = nblk();
  const float* mu_w = (const float*)d_in[pi++];
  const float* mu_b = (const float*)d_in[pi++];
  const float* lv_w = (const float*)d_in[pi++];
  const float* lv_b = (const float*)d_in[pi++];

  // workspace bump allocator (256B aligned)
  char* base = (char*)d_ws;
  size_t off = 0;
  auto alloc = [&](size_t bytes) -> char* {
    char* p = base + off;
    off = (off + bytes + 255) & ~(size_t)255;
    return p;
  };
  float* SKIP3 = (float*)alloc(49152ull * 64 * 4);    // x3  [4,12288,64]
  float* SKIP2 = (float*)alloc(12288ull * 128 * 4);   // x2  [4,3072,128]
  float* SKIP1 = (float*)alloc(3072ull * 256 * 4);    // x1  [4,768,256]
  float* ACTA  = (float*)alloc(49152ull * 128 * 4);   // ping
  float* ACTB  = (float*)alloc(49152ull * 128 * 4);   // pong
  float* X1    = (float*)alloc(49152ull * 128 * 4);   // L x0
  float* X2    = (float*)alloc(49152ull * 128 * 4);   // 2 L x1 - x0
  unsigned short* XS = (unsigned short*)alloc(49152ull * 384 * 2);  // bf16 A
  unsigned short* WF = (unsigned short*)alloc(1536ull * 512 * 2);   // bf16 W (swizzled)
  float* SS    = (float*)alloc(4096);                 // scale|shift (<=512 ch)

  const int V3 = 12288, V2 = 3072, V1 = 768, V0 = 192;

  // ---- Encoder 3 (lap3, no pool)
  run_cbr(stream, cols3, vals3, V3, x,    9,  32, E3.w1, E3.b1, E3.g1, E3.be1, X1, X2, XS, WF, SS, ACTA);
  run_cbr(stream, cols3, vals3, V3, ACTA, 32, 64, E3.w2, E3.b2, E3.g2, E3.be2, X1, X2, XS, WF, SS, SKIP3);
  // ---- Encoder 2 (pool 4, lap2)
  maxpool_kernel<<<gblk((long long)kB * V2 * 64), 256, 0, stream>>>(SKIP3, ACTA, V2, 64);
  run_cbr(stream, cols2, vals2, V2, ACTA, 64, 128, E2.w1, E2.b1, E2.g1, E2.be1, X1, X2, XS, WF, SS, ACTB);
  run_cbr(stream, cols2, vals2, V2, ACTB, 128, 128, E2.w2, E2.b2, E2.g2, E2.be2, X1, X2, XS, WF, SS, SKIP2);
  // ---- Encoder 1 (pool 4, lap1)
  maxpool_kernel<<<gblk((long long)kB * V1 * 128), 256, 0, stream>>>(SKIP2, ACTA, V1, 128);
  run_cbr(stream, cols1, vals1, V1, ACTA, 128, 256, E1.w1, E1.b1, E1.g1, E1.be1, X1, X2, XS, WF, SS, ACTB);
  run_cbr(stream, cols1, vals1, V1, ACTB, 256, 256, E1.w2, E1.b2, E1.g2, E1.be2, X1, X2, XS, WF, SS, SKIP1);
  // ---- Bottleneck (pool 4, lap0)
  maxpool_kernel<<<gblk((long long)kB * V0 * 256), 256, 0, stream>>>(SKIP1, ACTA, V0, 256);
  run_cbr(stream, cols0, vals0, V0, ACTA, 256, 512, Bb.w1, Bb.b1, Bb.g1, Bb.be1, X1, X2, XS, WF, SS, ACTB);
  run_cbr(stream, cols0, vals0, V0, ACTB, 512, 256, Bb.w2, Bb.b2, Bb.g2, Bb.be2, X1, X2, XS, WF, SS, ACTA);
  // ---- Decoder 1 (lap1)
  upconcat_kernel<<<gblk((long long)kB * V1 * 512), 256, 0, stream>>>(ACTA, SKIP1, ACTB, V1, 256, 256);
  run_cbr(stream, cols1, vals1, V1, ACTB, 512, 256, D1.w1, D1.b1, D1.g1, D1.be1, X1, X2, XS, WF, SS, ACTA);
  run_cbr(stream, cols1, vals1, V1, ACTA, 256, 128, D1.w2, D1.b2, D1.g2, D1.be2, X1, X2, XS, WF, SS, ACTB);
  // ---- Decoder 2 (lap2)
  upconcat_kernel<<<gblk((long long)kB * V2 * 256), 256, 0, stream>>>(ACTB, SKIP2, ACTA, V2, 128, 128);
  run_cbr(stream, cols2, vals2, V2, ACTA, 256, 128, D2.w1, D2.b1, D2.g1, D2.be1, X1, X2, XS, WF, SS, ACTB);
  run_cbr(stream, cols2, vals2, V2, ACTB, 128, 64, D2.w2, D2.b2, D2.g2, D2.be2, X1, X2, XS, WF, SS, ACTA);
  // ---- Decoder 3 (lap3)
  upconcat_kernel<<<gblk((long long)kB * V3 * 128), 256, 0, stream>>>(ACTA, SKIP3, ACTB, V3, 64, 64);
  run_cbr(stream, cols3, vals3, V3, ACTB, 128, 32, D3.w1, D3.b1, D3.g1, D3.be1, X1, X2, XS, WF, SS, ACTA);
  run_cbr(stream, cols3, vals3, V3, ACTA, 32, 16, D3.w2, D3.b2, D3.g2, D3.be2, X1, X2, XS, WF, SS, ACTB);
  // ---- Heads
  heads_kernel<<<gblk((long long)kB * V3), 256, 0, stream>>>(ACTB, mu_w, mu_b, lv_w, lv_b,
                                                             (float*)d_out, kB * V3);
}
